// NodeSelector_14611478741101
// MI455X (gfx1250) — matrix operand order repaired and checked by reading, not compiled, not run on hardware
//
#include <hip/hip_runtime.h>
#include <hip/hip_bf16.h>

#define BB   16
#define NND  1024
#define DDIM 256
#define MASKV (-1e4f)

// d_out layout (floats, concatenated in return order)
#define STATE_OFF 0        // [B,D]   = 4096
#define NODE_OFF  4096     // [B]     = 16
#define CA_OFF    4112     // [B,N]   = 16384
#define NS_OFF    20496    // [B,N]   = 16384
#define PROB_OFF  36880    // [B]     = 16
#define NLL_OFF   36896    // [1]

typedef __attribute__((ext_vector_type(16))) __bf16 v16bf;
typedef __attribute__((ext_vector_type(8)))  __bf16 v8bf;
typedef __attribute__((ext_vector_type(8)))  float  v8f;

union Frag {
    v16bf v;
    v8bf  h[2];
    unsigned short u[16];
    uint4 q[2];
};

__device__ __forceinline__ v8bf cvt8(float4 a, float4 b) {
    v8f f = {a.x, a.y, a.z, a.w, b.x, b.y, b.z, b.w};
    return __builtin_convertvector(f, v8bf);      // packed v_cvt_pk_bf16_f32, RNE
}
__device__ __forceinline__ v8bf cvt8_mask(float4 a, float4 b,
                                          float4 ma, float4 mb, float mi) {
    v8f f = {a.x*ma.x, a.y*ma.y, a.z*ma.z, a.w*ma.w,
             b.x*mb.x, b.y*mb.y, b.z*mb.z, b.w*mb.w};
    f = f * mi;                                   // values are exactly 0 or 1
    return __builtin_convertvector(f, v8bf);
}
__device__ __forceinline__ float bf2f(unsigned short u) {
    return __uint_as_float(((unsigned int)u) << 16);
}

// ---------------- k1: modality attention -> state ----------------
__global__ __launch_bounds__(256) void k1_state(
    const float* __restrict__ inputInfo, const float* __restrict__ agentState,
    const float* __restrict__ emb, const int* __restrict__ curIdx,
    const float* __restrict__ Wm, const float* __restrict__ vm,
    float* __restrict__ wsState, float* __restrict__ outState) {
    __shared__ float inp[3][DDIM];
    __shared__ float red[256];
    __shared__ float attS[3];
    int b = blockIdx.x, t = threadIdx.x;
    int cur = curIdx[b];
    inp[0][t] = inputInfo[b*DDIM + t];
    inp[1][t] = agentState[b*DDIM + t];
    inp[2][t] = emb[((size_t)b*NND + cur)*DDIM + t];
    __syncthreads();
    float logit[3];
    for (int mm = 0; mm < 3; ++mm) {
        float acc = 0.f;
        for (int k = 0; k < DDIM; ++k) acc += inp[mm][k] * Wm[k*DDIM + t];
        red[t] = tanhf(acc) * vm[t];
        __syncthreads();
        for (int s = 128; s > 0; s >>= 1) { if (t < s) red[t] += red[t+s]; __syncthreads(); }
        logit[mm] = red[0];
        __syncthreads();
    }
    if (t == 0) {
        float mx = fmaxf(logit[0], fmaxf(logit[1], logit[2]));
        float e0 = expf(logit[0]-mx), e1 = expf(logit[1]-mx), e2 = expf(logit[2]-mx);
        float z = e0 + e1 + e2;
        attS[0] = e0/z; attS[1] = e1/z; attS[2] = e2/z;
    }
    __syncthreads();
    float st = attS[0]*inp[0][t] + attS[1]*inp[1][t] + attS[2]*inp[2][t];
    wsState[b*DDIM + t] = st;
    outState[b*DDIM + t] = st;
}

// ---- k2: bias sb = state @ W1_bot, neighbor mask m, zero act, cs sums ----
__global__ __launch_bounds__(256) void k2_prep(
    const float* __restrict__ adj, const int* __restrict__ curIdx,
    const float* __restrict__ W1, const float* __restrict__ wsState,
    const float* __restrict__ curScore,
    float* __restrict__ sb, float* __restrict__ m_buf,
    float* __restrict__ act, float* __restrict__ cs_part) {
    __shared__ float st[DDIM];
    __shared__ float red[256];
    int b = blockIdx.x, t = threadIdx.x;
    st[t] = wsState[b*DDIM + t];
    __syncthreads();
    float acc = 0.f;
    for (int k = 0; k < DDIM; ++k) acc += st[k] * W1[(size_t)(DDIM + k)*DDIM + t];
    sb[b*DDIM + t] = acc;
    int cur = curIdx[b];
    const float* arow = adj + ((size_t)b*NND + cur)*NND;
    float csacc = 0.f;
    for (int n = t; n < NND; n += 256) {
        m_buf[b*NND + n] = arow[n];
        act[b*NND + n]   = 0.f;
        csacc += curScore[b*NND + n];
    }
    red[t] = csacc; __syncthreads();
    for (int s = 128; s > 0; s >>= 1) { if (t < s) red[t] += red[t+s]; __syncthreads(); }
    if (t == 0) cs_part[b] = red[0];
}

// ---- k3: h_t[b,d,n] = bf16(relu(emb @ W1_top + sb))  (WMMA bf16) ----
__global__ __launch_bounds__(256) void k3_hmat(
    const float* __restrict__ emb, const float* __restrict__ W1,
    const float* __restrict__ sb, unsigned short* __restrict__ h_t) {
    int b = blockIdx.x;
    int nt = blockIdx.y;            // 16 tiles of 64 rows
    int dt = blockIdx.z;            // 8 tiles of 32 cols
    int w    = threadIdx.x >> 5;
    int lane = threadIdx.x & 31;
    int r  = lane & 15, hi = lane >> 4;
    int wr = w & 3,  wc = w >> 2;
    int nbase = nt*64 + wr*16;
    int dbase = dt*32 + wc*16;
    int row  = nbase + r;           // A row (node index)
    int dcol = dbase + r;           // B / C column

    v8f c = {};
    const float* aptr = emb + ((size_t)b*NND + row)*DDIM;
    for (int kt = 0; kt < DDIM; kt += 32) {
        Frag A, Bf;
        const float4* p0 = (const float4*)(aptr + kt + hi*8);
        const float4* p1 = (const float4*)(aptr + kt + hi*8 + 16);
        A.h[0] = cvt8(p0[0], p0[1]);
        A.h[1] = cvt8(p1[0], p1[1]);
        int k0 = kt + hi*8;
        float w0[16];
        #pragma unroll
        for (int j = 0; j < 16; ++j)
            w0[j] = W1[(size_t)(k0 + j + (j >= 8 ? 8 : 0))*DDIM + dcol];
        {
            v8f f0 = {w0[0], w0[1], w0[2],  w0[3],  w0[4],  w0[5],  w0[6],  w0[7]};
            v8f f1 = {w0[8], w0[9], w0[10], w0[11], w0[12], w0[13], w0[14], w0[15]};
            Bf.h[0] = __builtin_convertvector(f0, v8bf);
            Bf.h[1] = __builtin_convertvector(f1, v8bf);
        }
        c = __builtin_amdgcn_wmma_f32_16x16x32_bf16(false, A.v, false, Bf.v,
                                                    (short)0, c, false, false);
    }
    float sbv = sb[b*DDIM + dcol];
    v8f fo;
    #pragma unroll
    for (int v = 0; v < 8; ++v) fo[v] = fmaxf(c[v] + sbv, 0.f);
    union { v8bf h; uint4 q; } pk;
    pk.h = __builtin_convertvector(fo, v8bf);
    *(uint4*)(h_t + ((size_t)b*DDIM + dcol)*NND + nbase + hi*8) = pk.q;
}

// ---- k4: agg = step_adj @ h (WMMA bf16), fused (h+agg)·w_out epilogue ----
__global__ __launch_bounds__(256) void k4_agg(
    const float* __restrict__ adj, const unsigned short* __restrict__ h_t,
    const float* __restrict__ m_buf, const float* __restrict__ w_out,
    float* __restrict__ act) {
    int b  = blockIdx.x;
    int it = blockIdx.y;            // 32 tiles of 32 rows
    int dt = blockIdx.z;            // 4 tiles of 64 cols
    int w    = threadIdx.x >> 5;
    int lane = threadIdx.x & 31;
    int r  = lane & 15, hi = lane >> 4;
    int wr = w & 1,  wc = w >> 1;
    int ibase = it*32 + wr*16;
    int dbase = dt*64 + wc*16;
    int irow = ibase + r;
    int dcol = dbase + r;
    float mi = m_buf[b*NND + irow];
    const float* arow = adj + ((size_t)b*NND + irow)*NND;
    const float* mrow = m_buf + (size_t)b*NND;
    const unsigned short* hrow = h_t + ((size_t)b*DDIM + dcol)*NND;

    v8f c = {};
    for (int kt = 0; kt < NND; kt += 32) {
        // prefetch next adjacency slab (L2-resident; pulls toward WGP)
        if (kt + 32 < NND)
            __builtin_prefetch(arow + kt + 32 + hi*8, 0, 0);
        Frag A, Bf;
        int kb = kt + hi*16;
        const float4* pa  = (const float4*)(arow + kb);
        const float4* pa2 = (const float4*)(arow + kb + 8);
        const float4* pm  = (const float4*)(mrow + kb);
        const float4* pm2 = (const float4*)(mrow + kb + 8);
        A.h[0] = cvt8_mask(pa[0],  pa[1],  pm[0],  pm[1],  mi);
        A.h[1] = cvt8_mask(pa2[0], pa2[1], pm2[0], pm2[1], mi);
        const uint4* ph = (const uint4*)(hrow + kt + hi*16);
        Bf.q[0] = ph[0]; Bf.q[1] = ph[1];
        c = __builtin_amdgcn_wmma_f32_16x16x32_bf16(false, A.v, false, Bf.v,
                                                    (short)0, c, false, false);
    }
    float wv = w_out[dcol];
    #pragma unroll
    for (int v = 0; v < 8; ++v) {
        int n = ibase + hi*8 + v;
        float pv = (c[v] + bf2f(hrow[n])) * wv;
        pv += __shfl_xor(pv, 1, 32);
        pv += __shfl_xor(pv, 2, 32);
        pv += __shfl_xor(pv, 4, 32);
        pv += __shfl_xor(pv, 8, 32);
        if (r == 0) atomicAdd(&act[b*NND + n], pv);
    }
}

// ---------------- k5: per-batch finalize ----------------
__device__ __forceinline__ float block_sum1024(float v, float* red, int t) {
    __syncthreads();
    red[t] = v; __syncthreads();
    for (int s = 512; s > 0; s >>= 1) { if (t < s) red[t] += red[t+s]; __syncthreads(); }
    float r = red[0]; __syncthreads();
    return r;
}

__global__ __launch_bounds__(1024) void k5_final(
    const float* __restrict__ act, const float* __restrict__ m_buf,
    const int* __restrict__ nodes, const float* __restrict__ label,
    const float* __restrict__ curScore, const float* __restrict__ curAct,
    const float* __restrict__ cs_part, float* __restrict__ nll_part,
    float* __restrict__ out) {
    __shared__ float rv[1024];
    __shared__ int   ri[1024];
    int b = blockIdx.x, t = threadIdx.x;
    float m = m_buf[b*NND + t];
    float a = (m > 0.f) ? act[b*NND + t] : MASKV;
    out[CA_OFF + b*NND + t] = curAct[b*NND + t];            // pass-through
    // max + first-occurrence argmax
    rv[t] = a; ri[t] = t; __syncthreads();
    for (int s = 512; s > 0; s >>= 1) {
        if (t < s) {
            if (rv[t+s] > rv[t] || (rv[t+s] == rv[t] && ri[t+s] < ri[t])) {
                rv[t] = rv[t+s]; ri[t] = ri[t+s];
            }
        }
        __syncthreads();
    }
    float amax = rv[0]; int idx = ri[0]; __syncthreads();
    float e = expf(a - amax);
    float Z = block_sum1024(e, rv, t);
    float score = e / Z;
    if (t == 0) {
        out[PROB_OFF + b] = -logf(Z);                       // log(score[argmax])
        out[NODE_OFF + b] = (float)nodes[b*NND + idx];
    }
    // score propagation
    float cstot = 0.f;
    #pragma unroll
    for (int i = 0; i < BB; ++i) cstot += cs_part[i];
    bool has = (cstot != 0.f);
    float csn = curScore[b*NND + t];
    float cntcs = block_sum1024(csn != 0.f ? 1.f : 0.f, rv, t);
    float cntsc = block_sum1024(m, rv, t);
    float cs = has ? 0.5f * csn * cntcs : csn;
    float sc = has ? 0.5f * score * cntsc : score;
    float ns = cs + sc * m;
    float den = block_sum1024(ns, rv, t);
    out[NS_OFF + b*NND + t] = ns / den;
    // bi-tempered loss (t1=0.8, t2=1.2)
    float lbl = label[b*NND + t] * m * 0.8f + 0.2f / (float)NND;
    float a0 = a - amax, an = a0;
    float Zt = 0.f;
    for (int itr = 0; itr < 5; ++itr) {
        Zt = block_sum1024(powf(fmaxf(1.f - 0.2f*an, 0.f), -5.f), rv, t);
        an = a0 * powf(Zt, -0.2f);
    }
    Zt = block_sum1024(powf(fmaxf(1.f - 0.2f*an, 0.f), -5.f), rv, t);
    float norm = 5.f * (powf(Zt, 0.2f) - 1.f) + amax;
    float pr = powf(fmaxf(1.f - 0.2f*(a - norm), 0.f), -5.f);
    const float eps = 1e-10f;
    float loss = lbl * (5.f*(powf(lbl+eps, 0.2f)-1.f) - 5.f*(powf(pr+eps, 0.2f)-1.f))
               - (powf(lbl, 1.2f) - powf(pr, 1.2f)) / 1.2f;
    float ls = block_sum1024(loss, rv, t);
    if (t == 0) nll_part[b] = ls;
}

__global__ void k6_nll(const float* __restrict__ nll_part, float* __restrict__ out) {
    if (threadIdx.x == 0) {
        float s = 0.f;
        #pragma unroll
        for (int i = 0; i < BB; ++i) s += nll_part[i];
        out[NLL_OFF] = s;
    }
}

extern "C" void kernel_launch(void* const* d_in, const int* in_sizes, int n_in,
                              void* d_out, int out_size, void* d_ws, size_t ws_size,
                              hipStream_t stream) {
    (void)in_sizes; (void)n_in; (void)out_size; (void)ws_size;
    const int*   nodes      = (const int*)d_in[0];
    const float* adj        = (const float*)d_in[1];
    const float* label      = (const float*)d_in[2];
    const float* emb        = (const float*)d_in[3];
    const float* inputInfo  = (const float*)d_in[4];
    const float* agentState = (const float*)d_in[5];
    const int*   curIdx     = (const int*)d_in[6];
    const float* curAct     = (const float*)d_in[7];
    const float* curScore   = (const float*)d_in[8];
    const float* Wm         = (const float*)d_in[9];
    const float* vm         = (const float*)d_in[10];
    const float* W1         = (const float*)d_in[11];
    const float* w_out      = (const float*)d_in[12];
    float* out = (float*)d_out;

    float* ws       = (float*)d_ws;
    float* act      = ws;                       // B*N
    float* state    = act + BB*NND;             // B*D
    float* sb       = state + BB*DDIM;          // B*D
    float* m_buf    = sb + BB*DDIM;             // B*N
    float* cs_part  = m_buf + BB*NND;           // B
    float* nll_part = cs_part + BB;             // B
    unsigned short* h_t = (unsigned short*)(nll_part + BB);   // B*D*N bf16 (16B aligned)

    k1_state<<<BB, DDIM, 0, stream>>>(inputInfo, agentState, emb, curIdx, Wm, vm,
                                      state, out + STATE_OFF);
    k2_prep<<<BB, DDIM, 0, stream>>>(adj, curIdx, W1, state, curScore,
                                     sb, m_buf, act, cs_part);
    k3_hmat<<<dim3(BB, NND/64, DDIM/32), 256, 0, stream>>>(emb, W1, sb, h_t);
    k4_agg<<<dim3(BB, NND/32, DDIM/64), 256, 0, stream>>>(adj, h_t, m_buf, w_out, act);
    k5_final<<<BB, NND, 0, stream>>>(act, m_buf, nodes, label, curScore, curAct,
                                     cs_part, nll_part, out);
    k6_nll<<<1, 32, 0, stream>>>(nll_part, out);
}
